// EnhancedDomainSeparationModule_53996328846062
// MI455X (gfx1250) — matrix operand order repaired and checked by reading, not compiled, not run on hardware
//
#include <hip/hip_runtime.h>
#include <hip/hip_bf16.h>
#include <math.h>

#define BB    8
#define CD    128
#define HH    48
#define WID   48
#define HW    2304
#define NN    2304
#define RD16  16

typedef __attribute__((ext_vector_type(16))) __bf16 v16bf;
typedef __attribute__((ext_vector_type(8)))  float  v8f;

static __device__ __forceinline__ unsigned short f2bfbits(float f) {
  unsigned u = __float_as_uint(f);
  unsigned r = (u + 0x7FFFu + ((u >> 16) & 1u)) >> 16;
  return (unsigned short)r;
}
static __device__ __forceinline__ __bf16 bf_from_bits(unsigned short s) {
  union { unsigned short u; __bf16 b; } cv; cv.u = s; return cv.b;
}
static __device__ __forceinline__ __bf16 f2bf(float f) { return bf_from_bits(f2bfbits(f)); }

static __device__ __forceinline__ float geluf(float v) {
  return 0.5f * v * (1.0f + erff(v * 0.7071067811865475f));
}
static __device__ __forceinline__ float activate(float v, int act) {
  if (act == 1) return geluf(v);
  if (act == 2) return 1.0f / (1.0f + __expf(-v));
  return v;
}

// ---------------- generic 1x1 conv (pointwise GEMM, memory-bound) ----------------
__global__ void k_conv1x1(const float* __restrict__ in, const float* __restrict__ w,
                          const float* __restrict__ bias, float* __restrict__ out,
                          int Ci, int Co, int spatial, int act) {
  int idx = blockIdx.x * blockDim.x + threadIdx.x;
  int total = BB * Co * spatial;
  if (idx >= total) return;
  int p = idx % spatial;
  int co = (idx / spatial) % Co;
  int b = idx / (spatial * Co);
  const float* ip = in + (size_t)b * Ci * spatial + p;
  const float* wp = w + (size_t)co * Ci;
  float acc = bias[co];
  for (int ci = 0; ci < Ci; ++ci) acc += ip[(size_t)ci * spatial] * wp[ci];
  out[idx] = activate(acc, act);
}

// ---------------- generic dense KxK conv (pad) on 48x48 ----------------
__global__ void k_convK(const float* __restrict__ in, const float* __restrict__ w,
                        const float* __restrict__ bias, float* __restrict__ out,
                        int Ci, int Co, int K, int pad, int act) {
  int idx = blockIdx.x * blockDim.x + threadIdx.x;
  int total = BB * Co * HW;
  if (idx >= total) return;
  int p = idx % HW;
  int co = (idx / HW) % Co;
  int b = idx / (HW * Co);
  int y = p / WID, x = p % WID;
  float acc = bias[co];
  for (int ci = 0; ci < Ci; ++ci) {
    const float* ip = in + (size_t)(b * Ci + ci) * HW;
    const float* wp = w + ((size_t)co * Ci + ci) * K * K;
    for (int ky = 0; ky < K; ++ky) {
      int iy = y + ky - pad;
      if (iy < 0 || iy >= HH) continue;
      for (int kx = 0; kx < K; ++kx) {
        int ix = x + kx - pad;
        if (ix < 0 || ix >= WID) continue;
        acc += ip[iy * WID + ix] * wp[ky * K + kx];
      }
    }
  }
  out[idx] = activate(acc, act);
}

// ---------------- depthwise 3x3, pad 1 ----------------
__global__ void k_dw3x3(const float* __restrict__ in, const float* __restrict__ w,
                        const float* __restrict__ bias, float* __restrict__ out, int C) {
  int idx = blockIdx.x * blockDim.x + threadIdx.x;
  int total = BB * C * HW;
  if (idx >= total) return;
  int p = idx % HW;
  int c = (idx / HW) % C;
  int b = idx / (HW * C);
  int y = p / WID, x = p % WID;
  const float* ip = in + (size_t)(b * C + c) * HW;
  const float* wp = w + (size_t)c * 9;
  float acc = bias[c];
  for (int ky = 0; ky < 3; ++ky) {
    int iy = y + ky - 1;
    if (iy < 0 || iy >= HH) continue;
    for (int kx = 0; kx < 3; ++kx) {
      int ix = x + kx - 1;
      if (ix < 0 || ix >= WID) continue;
      acc += ip[iy * WID + ix] * wp[ky * 3 + kx];
    }
  }
  out[idx] = acc;
}

// ---------------- BatchNorm: per-channel batch stats -> scale/shift ----------------
__global__ void k_bn_stats(const float* __restrict__ in, int C, int spatial,
                           const float* __restrict__ g, const float* __restrict__ bt,
                           float* __restrict__ scale, float* __restrict__ shift) {
  int c = blockIdx.x;
  float s = 0.0f, s2 = 0.0f;
  int n = BB * spatial;
  for (int i = threadIdx.x; i < n; i += blockDim.x) {
    int b = i / spatial, p = i % spatial;
    float v = in[((size_t)b * C + c) * spatial + p];
    s += v; s2 += v * v;
  }
  __shared__ float sh0[256], sh1[256];
  sh0[threadIdx.x] = s; sh1[threadIdx.x] = s2;
  __syncthreads();
  for (int o = 128; o > 0; o >>= 1) {
    if ((int)threadIdx.x < o) { sh0[threadIdx.x] += sh0[threadIdx.x + o]; sh1[threadIdx.x] += sh1[threadIdx.x + o]; }
    __syncthreads();
  }
  if (threadIdx.x == 0) {
    float fn = (float)n;
    float m = sh0[0] / fn;
    float v = sh1[0] / fn - m * m;
    float sc = g[c] * rsqrtf(v + 1e-5f);
    scale[c] = sc;
    shift[c] = bt[c] - m * sc;
  }
}

__global__ void k_bn_apply(const float* __restrict__ in, float* __restrict__ out,
                           const float* __restrict__ sc, const float* __restrict__ sh,
                           int C, int spatial, int act) {
  int idx = blockIdx.x * blockDim.x + threadIdx.x;
  int total = BB * C * spatial;
  if (idx >= total) return;
  int c = (idx / spatial) % C;
  out[idx] = activate(in[idx] * sc[c] + sh[c], act);
}

// ---------------- pack xr [B,16,HW] -> xg f32 [B,N,16] + bf16 Q/K and V copies ----------------
__global__ void k_pack(const float* __restrict__ xr, float* __restrict__ xg,
                       unsigned short* __restrict__ xf_bf, unsigned short* __restrict__ xg_bf) {
  int idx = blockIdx.x * blockDim.x + threadIdx.x;
  if (idx >= BB * NN * RD16) return;
  int c = idx % RD16;
  int nidx = (idx / RD16) % NN;
  int b = idx / (RD16 * NN);
  float v = xr[((size_t)b * RD16 + c) * HW + nidx];
  xg[idx] = v;
  unsigned short u = f2bfbits(v);
  xf_bf[idx] = u;
  xg_bf[idx] = u;
}

__global__ void k_unpack_add(const float* __restrict__ xg, const float* __restrict__ xr,
                             float* __restrict__ out) {
  int idx = blockIdx.x * blockDim.x + threadIdx.x;
  if (idx >= BB * RD16 * HW) return;
  int nidx = idx % HW;
  int c = (idx / HW) % RD16;
  int b = idx / (HW * RD16);
  out[idx] = xg[((size_t)b * NN + nidx) * RD16 + c] + xr[idx];
}

// ---------------- fused GCN layer: flash-attention-style adj@xg + linear + GELU + LN ----------
// adj@xg == (softmax(Q Kt / 4) @ V + 0.1*V_row) / 1.1   (softmax rowsum==1 -> L1 norm == 1.1)
// Q=K=xf(bf16), V=xg(bf16). wave32 per workgroup, one 16-row tile.
// K/V tiles are streamed into LDS with async global->LDS DMA (double-buffered, ASYNCcnt).
__global__ __launch_bounds__(32) void k_gcn(
    const unsigned short* __restrict__ xf_bf,
    const unsigned short* __restrict__ xgv_bf,
    const float* __restrict__ xg_in,
    const float* __restrict__ Wg, const float* __restrict__ bg,
    const float* __restrict__ lng, const float* __restrict__ lnb,
    float* __restrict__ xg_out, unsigned short* __restrict__ xg_out_bf) {
  __shared__ float sP[16 * 32];
  __shared__ unsigned short kst[2][32 * RD16];   // K tile double buffer (1KB each)
  __shared__ unsigned short vst[2][32 * RD16];   // V tile double buffer (1KB each)

  const int lane = threadIdx.x;
  const int hf = lane >> 4;       // lane half: C-layout rows r / r+8
  const int l16 = lane & 15;
  const int blk = blockIdx.x;
  const int b = blk / (NN / 16);
  const int i0 = (blk % (NN / 16)) * 16;

  const unsigned short* qb  = xf_bf  + (size_t)b * NN * RD16;
  const unsigned short* vb0 = xgv_bf + (size_t)b * NN * RD16;
  const __bf16 bz = bf_from_bits(0);

  // ---- async stage of one 32-row (1KB) tile: 2 x B128 per buffer, 512B per instruction ----
  auto stage_tile = [&](int jt, int which) {
    unsigned long long gK = (unsigned long long)(size_t)(qb  + (size_t)jt * 32 * RD16) + (unsigned)lane * 16u;
    unsigned long long gV = (unsigned long long)(size_t)(vb0 + (size_t)jt * 32 * RD16) + (unsigned)lane * 16u;
    unsigned lK = (unsigned)(size_t)(&kst[which][0]) + (unsigned)lane * 16u;
    unsigned lV = (unsigned)(size_t)(&vst[which][0]) + (unsigned)lane * 16u;
    asm volatile("global_load_async_to_lds_b128 %0, %1, off" :: "v"(lK), "v"(gK) : "memory");
    asm volatile("global_load_async_to_lds_b128 %0, %1, off" :: "v"(lV), "v"(gV) : "memory");
    gK += 512u; gV += 512u; lK += 512u; lV += 512u;
    asm volatile("global_load_async_to_lds_b128 %0, %1, off" :: "v"(lK), "v"(gK) : "memory");
    asm volatile("global_load_async_to_lds_b128 %0, %1, off" :: "v"(lV), "v"(gV) : "memory");
  };

  // A tile (Q rows i0..i0+15): 16-bit A layout: lane row = l16, K-octet by half; K=16..31 pad 0
  v16bf Aq;
  {
    union { uint4 u; unsigned short s[8]; } cv;
    cv.u = *(const uint4*)(qb + (size_t)(i0 + l16) * RD16 + hf * 8);
#pragma unroll
    for (int e = 0; e < 8; ++e) Aq[e] = bf_from_bits(cv.s[e]);
#pragma unroll
    for (int e = 8; e < 16; ++e) Aq[e] = bz;
  }

  v8f acc = {};
  float mrun[8], lrun[8];
#pragma unroll
  for (int r = 0; r < 8; ++r) { mrun[r] = -1e30f; lrun[r] = 0.0f; }

  const int NT = NN / 32;        // 72 column tiles
  stage_tile(0, 0);

  for (int t = 0; t < NT; ++t) {
    const int cur = t & 1;
    if (t + 1 < NT) {
      stage_tile(t + 1, cur ^ 1);
      asm volatile("s_wait_asynccnt 0x4" ::: "memory");   // current tile's 4 DMAs retired
    } else {
      asm volatile("s_wait_asynccnt 0x0" ::: "memory");
    }

    const unsigned short* kb = &kst[cur][0];
    const unsigned short* vb = &vst[cur][0];

    // B tiles (Kt cols): B layout lane col=l16, elem e<8 -> K=8*hf+e (elems 8..15 = K>=16 zero pad)
    v16bf Bk0, Bk1;
#pragma unroll
    for (int e = 0; e < 16; ++e) {
      Bk0[e] = (e >= 8) ? bz : bf_from_bits(kb[l16 * RD16 + 8 * hf + e]);
      Bk1[e] = (e >= 8) ? bz : bf_from_bits(kb[(16 + l16) * RD16 + 8 * hf + e]);
    }
    v8f zc = {};
    v8f S0 = __builtin_amdgcn_wmma_f32_16x16x32_bf16(false, Aq, false, Bk0, (short)0, zc, false, false);
    v8f S1 = __builtin_amdgcn_wmma_f32_16x16x32_bf16(false, Aq, false, Bk1, (short)0, zc, false, false);

    // online softmax update per row (row = r + 8*hf, spread over 16-lane halves)
    float p0[8], p1[8];
#pragma unroll
    for (int r = 0; r < 8; ++r) {
      float a = S0[r] * 0.25f;        // / sqrt(RD)
      float c = S1[r] * 0.25f;
      float mx = fmaxf(a, c);
      mx = fmaxf(mx, __shfl_xor(mx, 1));
      mx = fmaxf(mx, __shfl_xor(mx, 2));
      mx = fmaxf(mx, __shfl_xor(mx, 4));
      mx = fmaxf(mx, __shfl_xor(mx, 8));
      float nm = fmaxf(mrun[r], mx);
      float sc = __expf(mrun[r] - nm);
      float e0 = __expf(a - nm);
      float e1 = __expf(c - nm);
      float se = e0 + e1;
      se += __shfl_xor(se, 1);
      se += __shfl_xor(se, 2);
      se += __shfl_xor(se, 4);
      se += __shfl_xor(se, 8);
      lrun[r] = lrun[r] * sc + se;
      mrun[r] = nm;
      acc[r] *= sc;
      p0[r] = e0; p1[r] = e1;
    }

    // transpose P (C layout) -> A layout via LDS
    __syncthreads();
#pragma unroll
    for (int r = 0; r < 8; ++r) {
      sP[(r + 8 * hf) * 32 + l16] = p0[r];
      sP[(r + 8 * hf) * 32 + 16 + l16] = p1[r];
    }
    __syncthreads();

    v16bf Ap, Bv;
    {
      const int k0 = hf * 8;
#pragma unroll
      for (int e = 0; e < 8; ++e) Ap[e] = f2bf(sP[l16 * 32 + k0 + e]);
#pragma unroll
      for (int e = 0; e < 8; ++e) Ap[8 + e] = f2bf(sP[l16 * 32 + 16 + k0 + e]);
#pragma unroll
      for (int e = 0; e < 16; ++e) Bv[e] = bf_from_bits(vb[((e & 7) + 8 * hf + 16 * (e >> 3)) * RD16 + l16]);
    }
    acc = __builtin_amdgcn_wmma_f32_16x16x32_bf16(false, Ap, false, Bv, (short)0, acc, false, false);
    __syncthreads();   // sP reads done before next iteration rewrites it
  }

  // sup = (acc / l + 0.1 * xg_row) / 1.1
  const float* resp = xg_in + ((size_t)b * NN + i0) * RD16;
  float sup[8];
#pragma unroll
  for (int r = 0; r < 8; ++r) {
    int row = r + 8 * hf;
    float v = acc[r] / lrun[r];
    sup[r] = (v + 0.1f * resp[(size_t)row * RD16 + l16]) * (1.0f / 1.1f);
  }
  __syncthreads();
#pragma unroll
  for (int r = 0; r < 8; ++r) sP[(r + 8 * hf) * 16 + l16] = sup[r];
  __syncthreads();

  // linear 16x16 via WMMA (K padded to 32)
  v16bf As, Bw;
  {
    const int k0 = hf * 8;
#pragma unroll
    for (int e = 0; e < 8; ++e) As[e] = f2bf(sP[l16 * 16 + k0 + e]);
#pragma unroll
    for (int e = 8; e < 16; ++e) As[e] = bz;
#pragma unroll
    for (int e = 0; e < 16; ++e) Bw[e] = (e >= 8) ? bz : f2bf(Wg[(8 * hf + e) * 16 + l16]); // W stored [in,out]
  }
  v8f zc2 = {};
  v8f Y = __builtin_amdgcn_wmma_f32_16x16x32_bf16(false, As, false, Bw, (short)0, zc2, false, false);

  // +bias, GELU, LayerNorm over 16 features (cross-lane within halves), write f32+bf16
  float bb_ = bg[l16], gg = lng[l16], bt_ = lnb[l16];
#pragma unroll
  for (int r = 0; r < 8; ++r) {
    float v = geluf(Y[r] + bb_);
    float s = v;
    s += __shfl_xor(s, 1); s += __shfl_xor(s, 2); s += __shfl_xor(s, 4); s += __shfl_xor(s, 8);
    float mean = s * (1.0f / 16.0f);
    float d = v - mean;
    float q = d * d;
    q += __shfl_xor(q, 1); q += __shfl_xor(q, 2); q += __shfl_xor(q, 4); q += __shfl_xor(q, 8);
    float var = q * (1.0f / 16.0f);
    float o = d * rsqrtf(var + 1e-5f) * gg + bt_;
    int row = r + 8 * hf;
    size_t oi = ((size_t)b * NN + i0 + row) * RD16 + l16;
    xg_out[oi] = o;
    xg_out_bf[oi] = f2bfbits(o);
  }
}

// ---------------- feature gate: 1x1 conv(16->2) + softmax + gated combine ----------------
__global__ void k_fgate(const float* __restrict__ d, const float* __restrict__ g,
                        const float* __restrict__ w2, const float* __restrict__ b2,
                        float* __restrict__ refined) {
  int idx = blockIdx.x * blockDim.x + threadIdx.x;
  if (idx >= BB * HW) return;
  int p = idx % HW;
  int b = idx / HW;
  const float* gp = g + (size_t)b * 16 * HW + p;
  float z0 = b2[0], z1 = b2[1];
  for (int c = 0; c < 16; ++c) {
    float gv = gp[(size_t)c * HW];
    z0 += w2[c] * gv;
    z1 += w2[16 + c] * gv;
  }
  float m = fmaxf(z0, z1);
  float e0 = __expf(z0 - m), e1 = __expf(z1 - m);
  float inv = 1.0f / (e0 + e1);
  float g0 = e0 * inv, g1 = e1 * inv;
  const float* dp = d + (size_t)b * 32 * HW + p;
  float* rp = refined + (size_t)b * 16 * HW + p;
  for (int c = 0; c < 16; ++c)
    rp[(size_t)c * HW] = g0 * dp[(size_t)c * HW] - g1 * dp[(size_t)(16 + c) * HW];
}

// ---------------- pyramid pooling / channel attention ----------------
__global__ void k_pool(const float* __restrict__ in, float* __restrict__ out, int s) {
  int idx = blockIdx.x * blockDim.x + threadIdx.x;
  int total = BB * CD * s * s;
  if (idx >= total) return;
  int sx = idx % s;
  int sy = (idx / s) % s;
  int c = (idx / (s * s)) % CD;
  int b = idx / (s * s * CD);
  int r = HH / s;
  const float* ip = in + (size_t)(b * CD + c) * HW;
  float sum = 0.0f;
  for (int yy = 0; yy < r; ++yy)
    for (int xx = 0; xx < r; ++xx)
      sum += ip[(sy * r + yy) * WID + sx * r + xx];
  out[idx] = sum / (float)(r * r);
}

__global__ void k_ca_resize(const float* __restrict__ a, float* __restrict__ ca, int s, int add) {
  int idx = blockIdx.x * blockDim.x + threadIdx.x;
  if (idx >= BB * CD * HW) return;
  int p = idx % HW;
  int c = (idx / HW) % CD;
  int b = idx / (HW * CD);
  float val;
  if (s == 1) {
    val = a[(size_t)b * CD + c];
  } else {
    int y = p / WID, x = p % WID;
    float fy = y * (float)(s - 1) / 47.0f;
    float fx = x * (float)(s - 1) / 47.0f;
    int y0 = (int)floorf(fy); if (y0 > s - 2) y0 = s - 2; if (y0 < 0) y0 = 0;
    int x0 = (int)floorf(fx); if (x0 > s - 2) x0 = s - 2; if (x0 < 0) x0 = 0;
    float wy = fy - (float)y0, wx = fx - (float)x0;
    const float* ap = a + (size_t)(b * CD + c) * s * s;
    float v00 = ap[y0 * s + x0],       v01 = ap[y0 * s + x0 + 1];
    float v10 = ap[(y0 + 1) * s + x0], v11 = ap[(y0 + 1) * s + x0 + 1];
    val = (v00 * (1.0f - wy) + v10 * wy) * (1.0f - wx) + (v01 * (1.0f - wy) + v11 * wy) * wx;
  }
  if (add) ca[idx] += val; else ca[idx] = val;
}

__global__ void k_mulca(float* __restrict__ out, const float* __restrict__ ca) {
  int idx = blockIdx.x * blockDim.x + threadIdx.x;
  if (idx >= BB * CD * HW) return;
  out[idx] = out[idx] * (ca[idx] * (1.0f / 3.0f));
}

// ---------------- spatial statistics over channels ----------------
__global__ void k_spstats(const float* __restrict__ in, float* __restrict__ sp) {
  int idx = blockIdx.x * blockDim.x + threadIdx.x;
  if (idx >= BB * HW) return;
  int p = idx % HW;
  int b = idx / HW;
  const float* ip = in + (size_t)b * CD * HW + p;
  float s = 0.0f, s2 = 0.0f, mx = -3.4e38f, mn = 3.4e38f;
  for (int c = 0; c < CD; ++c) {
    float v = ip[(size_t)c * HW];
    s += v; s2 += v * v;
    mx = fmaxf(mx, v); mn = fminf(mn, v);
  }
  float mean = s / (float)CD;
  float var = (s2 - (float)CD * mean * mean) / (float)(CD - 1);  // unbiased (torch.std)
  float sd = sqrtf(fmaxf(var, 0.0f));
  float* op = sp + (size_t)b * 4 * HW + p;
  op[0] = mean; op[HW] = mx; op[2 * HW] = mn; op[3 * HW] = sd;
}

__global__ void k_final(const float* __restrict__ outb, const float* __restrict__ sp1,
                        const float* __restrict__ xin, float* __restrict__ dout) {
  int idx = blockIdx.x * blockDim.x + threadIdx.x;
  if (idx >= BB * CD * HW) return;
  int p = idx % HW;
  int b = idx / (HW * CD);
  dout[idx] = outb[idx] * sp1[(size_t)b * HW + p] + xin[idx];
}

// ======================================================================================
extern "C" void kernel_launch(void* const* d_in, const int* in_sizes, int n_in,
                              void* d_out, int out_size, void* d_ws, size_t ws_size,
                              hipStream_t stream) {
  (void)in_sizes; (void)n_in; (void)out_size; (void)ws_size;
  const float* x = (const float*)d_in[0];
#define P(i) ((const float*)d_in[i])

  char* wsb = (char*)d_ws;
  size_t off = 0;
  auto carve = [&](size_t bytes) -> char* {
    char* p = wsb + off;
    off += (bytes + 255) & ~(size_t)255;
    return p;
  };
  float* f0  = (float*)carve((size_t)BB * 16 * HW * 4);   // conv tmp (16 ch)
  float* f1  = (float*)carve((size_t)BB * 16 * HW * 4);   // xr
  float* f2  = (float*)carve((size_t)BB * NN * 16 * 4);   // xg ping
  float* f3  = (float*)carve((size_t)BB * NN * 16 * 4);   // xg pong
  float* f4  = (float*)carve((size_t)BB * 16 * HW * 4);   // xg_img / refined
  float* f5a = (float*)carve((size_t)BB * 32 * HW * 4);   // ds tmp a / d
  float* f5b = (float*)carve((size_t)BB * 32 * HW * 4);   // ds tmp b
  float* f6  = (float*)carve((size_t)BB * 16 * HW * 4);   // gate feature
  float* fo  = (float*)carve((size_t)BB * CD * HW * 4);   // rb tmp -> CA accumulator
  float* fca = (float*)carve((size_t)BB * CD * HW * 4);   // "out"
  float* poolb = (float*)carve((size_t)BB * CD * 16 * 4);
  float* attn1 = (float*)carve((size_t)BB * 8 * 16 * 4);
  float* attn2 = (float*)carve((size_t)BB * CD * 16 * 4);
  float* sp  = (float*)carve((size_t)BB * 4 * HW * 4);
  float* spt = (float*)carve((size_t)BB * 2 * HW * 4);
  float* sp1 = (float*)carve((size_t)BB * 1 * HW * 4);
  float* bnsc = (float*)carve(128 * 4);
  float* bnsh = (float*)carve(128 * 4);
  unsigned short* xfbf  = (unsigned short*)carve((size_t)BB * NN * 16 * 2);
  unsigned short* xgbfA = (unsigned short*)carve((size_t)BB * NN * 16 * 2);
  unsigned short* xgbfB = (unsigned short*)carve((size_t)BB * NN * 16 * 2);

  auto gsz = [](int total) { return dim3((unsigned)((total + 255) / 256)); };
  const dim3 blk256(256);

  // ---- reducer ----
  k_conv1x1<<<gsz(BB * 16 * HW), blk256, 0, stream>>>(x, P(1), P(2), f0, 128, 16, HW, 0);
  k_bn_stats<<<16, blk256, 0, stream>>>(f0, 16, HW, P(3), P(4), bnsc, bnsh);
  k_bn_apply<<<gsz(BB * 16 * HW), blk256, 0, stream>>>(f0, f0, bnsc, bnsh, 16, HW, 1);
  k_dw3x3<<<gsz(BB * 16 * HW), blk256, 0, stream>>>(f0, P(5), P(6), f1, 16);
  k_bn_stats<<<16, blk256, 0, stream>>>(f1, 16, HW, P(7), P(8), bnsc, bnsh);
  k_bn_apply<<<gsz(BB * 16 * HW), blk256, 0, stream>>>(f1, f1, bnsc, bnsh, 16, HW, 1);  // xr=f1

  // ---- pack + 3 fused GCN (flash-attention) layers ----
  k_pack<<<gsz(BB * NN * 16), blk256, 0, stream>>>(f1, f2, xfbf, xgbfA);
  dim3 ggrid(BB * (NN / 16));
  k_gcn<<<ggrid, 32, 0, stream>>>(xfbf, xgbfA, f2, P(9),  P(10), P(11), P(12), f3, xgbfB);
  k_gcn<<<ggrid, 32, 0, stream>>>(xfbf, xgbfB, f3, P(13), P(14), P(15), P(16), f2, xgbfA);
  k_gcn<<<ggrid, 32, 0, stream>>>(xfbf, xgbfA, f2, P(17), P(18), P(19), P(20), f3, xgbfB);
  k_unpack_add<<<gsz(BB * 16 * HW), blk256, 0, stream>>>(f3, f1, f4);

  // ---- domain separator ----
  k_conv1x1<<<gsz(BB * 32 * HW), blk256, 0, stream>>>(f4, P(21), P(22), f5a, 16, 32, HW, 0);
  k_bn_stats<<<32, blk256, 0, stream>>>(f5a, 32, HW, P(23), P(24), bnsc, bnsh);
  k_bn_apply<<<gsz(BB * 32 * HW), blk256, 0, stream>>>(f5a, f5a, bnsc, bnsh, 32, HW, 1);
  k_convK<<<gsz(BB * 32 * HW), blk256, 0, stream>>>(f5a, P(25), P(26), f5b, 32, 32, 3, 1, 0);
  k_bn_stats<<<32, blk256, 0, stream>>>(f5b, 32, HW, P(27), P(28), bnsc, bnsh);
  k_bn_apply<<<gsz(BB * 32 * HW), blk256, 0, stream>>>(f5b, f5b, bnsc, bnsh, 32, HW, 1);
  k_conv1x1<<<gsz(BB * 32 * HW), blk256, 0, stream>>>(f5b, P(29), P(30), f5a, 32, 32, HW, 0); // d

  // ---- feature gate ----
  k_conv1x1<<<gsz(BB * 16 * HW), blk256, 0, stream>>>(f5a, P(31), P(32), f6, 32, 16, HW, 0);
  k_bn_stats<<<16, blk256, 0, stream>>>(f6, 16, HW, P(33), P(34), bnsc, bnsh);
  k_bn_apply<<<gsz(BB * 16 * HW), blk256, 0, stream>>>(f6, f6, bnsc, bnsh, 16, HW, 1);
  k_fgate<<<gsz(BB * HW), blk256, 0, stream>>>(f5a, f6, P(35), P(36), f4);  // refined -> f4

  // ---- rebuilder ----
  k_conv1x1<<<gsz(BB * CD * HW), blk256, 0, stream>>>(f4, P(37), P(38), fo, 16, 128, HW, 0);
  k_bn_stats<<<128, blk256, 0, stream>>>(fo, 128, HW, P(39), P(40), bnsc, bnsh);
  k_bn_apply<<<gsz(BB * CD * HW), blk256, 0, stream>>>(fo, fo, bnsc, bnsh, 128, HW, 1);
  k_dw3x3<<<gsz(BB * CD * HW), blk256, 0, stream>>>(fo, P(41), P(42), fca, 128);
  k_bn_stats<<<128, blk256, 0, stream>>>(fca, 128, HW, P(43), P(44), bnsc, bnsh);
  k_bn_apply<<<gsz(BB * CD * HW), blk256, 0, stream>>>(fca, fca, bnsc, bnsh, 128, HW, 1); // out=fca

  // ---- pyramid channel attention (accumulate into fo) ----
  for (int i = 0; i < 3; ++i) {
    int s = (i == 0) ? 1 : (i == 1) ? 2 : 4;
    k_pool<<<gsz(BB * CD * s * s), blk256, 0, stream>>>(fca, poolb, s);
    k_conv1x1<<<gsz(BB * 8 * s * s), blk256, 0, stream>>>(poolb, P(45 + 4 * i), P(46 + 4 * i),
                                                          attn1, 128, 8, s * s, 1);
    k_conv1x1<<<gsz(BB * CD * s * s), blk256, 0, stream>>>(attn1, P(47 + 4 * i), P(48 + 4 * i),
                                                           attn2, 8, 128, s * s, 2);
    k_ca_resize<<<gsz(BB * CD * HW), blk256, 0, stream>>>(attn2, fo, s, (i == 0) ? 0 : 1);
  }
  k_mulca<<<gsz(BB * CD * HW), blk256, 0, stream>>>(fca, fo);

  // ---- spatial-stat gating ----
  k_spstats<<<gsz(BB * HW), blk256, 0, stream>>>(fca, sp);
  k_convK<<<gsz(BB * 2 * HW), blk256, 0, stream>>>(sp, P(57), P(58), spt, 4, 2, 7, 3, 0);
  k_bn_stats<<<2, blk256, 0, stream>>>(spt, 2, HW, P(59), P(60), bnsc, bnsh);
  k_bn_apply<<<gsz(BB * 2 * HW), blk256, 0, stream>>>(spt, spt, bnsc, bnsh, 2, HW, 1);
  k_convK<<<gsz(BB * 1 * HW), blk256, 0, stream>>>(spt, P(61), P(62), sp1, 2, 1, 7, 3, 2);

  k_final<<<gsz(BB * CD * HW), blk256, 0, stream>>>(fca, sp1, x, (float*)d_out);
#undef P
}
